// RelationalSlotAttention_29789893165550
// MI455X (gfx1250) — matrix operand order repaired and checked by reading, not compiled, not run on hardware
//
#include <hip/hip_runtime.h>
#include <hip/hip_bf16.h>

typedef __attribute__((ext_vector_type(16))) __bf16 bf16x16;
typedef __attribute__((ext_vector_type(8)))  float  f32x8;

#define B_    32
#define N_    4096
#define D_    256
#define NS_   8
#define HID_  128
#define ITERS_ 3

__device__ __forceinline__ float wred(float v) {
#pragma unroll
  for (int m = 16; m >= 1; m >>= 1) v += __shfl_xor(v, m, 32);
  return v;
}

// ---------------------------------------------------------------------------
// prep: bf16-cast Wk/Wv, init slots = mu + noise*exp(sigma)
// ---------------------------------------------------------------------------
__global__ void prep_kernel(const float* __restrict__ Wk, const float* __restrict__ Wv,
                            const float* __restrict__ smu, const float* __restrict__ ssig,
                            const float* __restrict__ noise,
                            __bf16* __restrict__ wk_bf, __bf16* __restrict__ wv_bf,
                            float* __restrict__ slots) {
  int idx = blockIdx.x * 256 + threadIdx.x;
  if (idx < D_ * D_) { wk_bf[idx] = (__bf16)Wk[idx]; return; }
  idx -= D_ * D_;
  if (idx < D_ * D_) { wv_bf[idx] = (__bf16)Wv[idx]; return; }
  idx -= D_ * D_;
  if (idx < B_ * NS_ * D_) {
    int d = idx & (D_ - 1);
    slots[idx] = smu[d] + noise[idx] * __expf(ssig[d]);
  }
}

// ---------------------------------------------------------------------------
// Fused LayerNorm(inputs) -> bf16 LDS tile -> WMMA K/V projection.
// K stored row-major [b][j][d] (dots reads K rows contiguously).
// V stored transposed [b][d][j] (updates reads V columns contiguously).
// ---------------------------------------------------------------------------
__global__ void ln_kv_kernel(const float* __restrict__ inputs,
                             const float* __restrict__ ln_g, const float* __restrict__ ln_b,
                             const __bf16* __restrict__ wk_bf, const __bf16* __restrict__ wv_bf,
                             __bf16* __restrict__ k_bf, __bf16* __restrict__ v_t) {
  __shared__ __bf16 xs[16][D_];
  const int tid = threadIdx.x;
  const int lane = tid & 31, wave = tid >> 5;
  const long rowbase = (long)blockIdx.x * 16;
  const int bb = (int)(rowbase >> 12);           // N_ == 4096
  const int jloc0 = (int)(rowbase & (N_ - 1));

#pragma unroll
  for (int rr = 0; rr < 2; ++rr) {
    const int lr = wave * 2 + rr;
    const float* xrow = inputs + (rowbase + lr) * D_;
    float xv[8];
    float s = 0.f;
#pragma unroll
    for (int t = 0; t < 8; ++t) { xv[t] = xrow[lane + 32 * t]; s += xv[t]; }
    const float mu = wred(s) * (1.0f / D_);
    float vs = 0.f;
#pragma unroll
    for (int t = 0; t < 8; ++t) { float d = xv[t] - mu; vs += d * d; }
    const float inv = rsqrtf(wred(vs) * (1.0f / D_) + 1e-5f);
#pragma unroll
    for (int t = 0; t < 8; ++t) {
      const int e = lane + 32 * t;
      xs[lr][e] = (__bf16)((xv[t] - mu) * inv * ln_g[e] + ln_b[e]);
    }
  }
  __syncthreads();

  const int m = lane & 15;
  const int kb = (lane >> 4) * 8;
  const int kb2 = (lane >> 4) * 16;
  for (int which = 0; which < 2; ++which) {
    const __bf16* W = which ? wv_bf : wk_bf;
    for (int t = wave; t < 16; t += 8) {
      const int ecol = t * 16 + m;
      f32x8 acc = {};
#pragma unroll
      for (int kc = 0; kc < 8; ++kc) {
        const int k0 = kc * 32;
        bf16x16 a, bm;
#pragma unroll
        for (int p = 0; p < 8; ++p) {
          a[p]     = xs[m][k0 + kb2 + p];
          a[8 + p] = xs[m][k0 + kb2 + 8 + p];
        }
        bm = *(const bf16x16*)(W + (long)ecol * D_ + k0 + kb2);
        acc = __builtin_amdgcn_wmma_f32_16x16x32_bf16(false, a, false, bm, (short)0, acc,
                                                      false, false);
      }
      if (which == 0) {
#pragma unroll
        for (int r = 0; r < 8; ++r) {
          const long gm = rowbase + r + (lane >> 4) * 8;
          k_bf[gm * D_ + t * 16 + m] = (__bf16)acc[r];
        }
      } else {
#pragma unroll
        for (int r = 0; r < 8; ++r) {
          const int jl = jloc0 + r + (lane >> 4) * 8;
          v_t[((long)bb * D_ + t * 16 + m) * N_ + jl] = (__bf16)acc[r];
        }
      }
    }
  }
}

// ---------------------------------------------------------------------------
// q = LN(slots) @ Wq^T  (padded to 16 rows, stored bf16 for WMMA A)
// ---------------------------------------------------------------------------
__global__ void qproj_kernel(const float* __restrict__ slots,
                             const float* __restrict__ ln_s_g, const float* __restrict__ ln_s_b,
                             const float* __restrict__ Wq, __bf16* __restrict__ q_bf) {
  __shared__ float sln[NS_][D_];
  const int b = blockIdx.x;
  const int tid = threadIdx.x, lane = tid & 31, wave = tid >> 5;
  {
    const float* srow = slots + ((long)b * NS_ + wave) * D_;
    float xv[8];
    float s = 0.f;
#pragma unroll
    for (int t = 0; t < 8; ++t) { xv[t] = srow[lane + 32 * t]; s += xv[t]; }
    const float mu = wred(s) * (1.0f / D_);
    float vs = 0.f;
#pragma unroll
    for (int t = 0; t < 8; ++t) { float d = xv[t] - mu; vs += d * d; }
    const float inv = rsqrtf(wred(vs) * (1.0f / D_) + 1e-5f);
#pragma unroll
    for (int t = 0; t < 8; ++t) {
      const int e = lane + 32 * t;
      sln[wave][e] = (xv[t] - mu) * inv * ln_s_g[e] + ln_s_b[e];
    }
  }
  __syncthreads();
  for (int o = tid; o < 16 * D_; o += 256) {
    const int i = o >> 8, e = o & 255;
    float val = 0.f;
    if (i < NS_) {
      const float* wrow = Wq + (long)e * D_;
      for (int d = 0; d < D_; ++d) val += sln[i][d] * wrow[d];
    }
    q_bf[(long)b * 16 * D_ + o] = (__bf16)val;
  }
}

// ---------------------------------------------------------------------------
// dots[b,i,j] = (q . k) * scale   via WMMA.
// q tile staged into LDS with CDNA5 async-to-LDS (ASYNCcnt) path.
// ---------------------------------------------------------------------------
__global__ void dots_kernel(const __bf16* __restrict__ q_bf, const __bf16* __restrict__ k_bf,
                            float* __restrict__ dots) {
  __shared__ __bf16 qs[16 * D_];
  const int tid = threadIdx.x;
  const int b = blockIdx.x >> 5;
  const int grp = blockIdx.x & 31;
  {
    // 8KB q tile: 512 x b128 async transfers, 2 per thread; no VGPR round-trip.
    const __bf16* src = q_bf + (long)b * 16 * D_;
#pragma unroll
    for (int t2 = 0; t2 < 2; ++t2) {
      const int t = tid + 256 * t2;
      const unsigned ldsa = (unsigned)(size_t)(&qs[t * 8]);
      const unsigned long long ga = (unsigned long long)(size_t)(src + t * 8);
      asm volatile("global_load_async_to_lds_b128 %0, %1, off"
                   :: "v"(ldsa), "v"(ga)
                   : "memory");
    }
    asm volatile("s_wait_asynccnt 0x0" ::: "memory");
  }
  __syncthreads();
  const int lane = tid & 31, wave = tid >> 5;
  const int jbase = (grp * 8 + wave) * 16;
  const int m = lane & 15, kb = (lane >> 4) * 8, kb2 = (lane >> 4) * 16;
  const long krow = ((long)b * N_ + jbase + m) * D_;
  f32x8 acc = {};
#pragma unroll
  for (int kc = 0; kc < 8; ++kc) {
    const int k0 = kc * 32;
    bf16x16 a, bm;
#pragma unroll
    for (int p = 0; p < 8; ++p) {
      a[p]     = qs[m * D_ + k0 + kb2 + p];
      a[8 + p] = qs[m * D_ + k0 + kb2 + 8 + p];
    }
    bm = *(const bf16x16*)(k_bf + krow + k0 + kb2);
    acc = __builtin_amdgcn_wmma_f32_16x16x32_bf16(false, a, false, bm, (short)0, acc,
                                                  false, false);
  }
  if (lane < 16) {
#pragma unroll
    for (int r = 0; r < NS_; ++r)
      dots[((long)b * NS_ + r) * N_ + jbase + m] = acc[r] * 0.0625f;
  }
}

// ---------------------------------------------------------------------------
// softmax over slots axis (NS=8) + EPS; write to attn buffer (in place) + d_out
// ---------------------------------------------------------------------------
__global__ void softmax_kernel(float* __restrict__ dots, float* __restrict__ attn_out) {
  const long gj = (long)blockIdx.x * 256 + threadIdx.x;  // over B*N
  const int b = (int)(gj / N_);
  const int j = (int)(gj % N_);
  float v[NS_];
  float mx = -1e30f;
#pragma unroll
  for (int i = 0; i < NS_; ++i) {
    v[i] = dots[((long)b * NS_ + i) * N_ + j];
    mx = fmaxf(mx, v[i]);
  }
  float s = 0.f;
#pragma unroll
  for (int i = 0; i < NS_; ++i) { v[i] = __expf(v[i] - mx); s += v[i]; }
  const float invs = 1.0f / s;
#pragma unroll
  for (int i = 0; i < NS_; ++i) {
    const float a = v[i] * invs + 1e-8f;
    dots[((long)b * NS_ + i) * N_ + j] = a;
    attn_out[((long)b * NS_ + i) * N_ + j] = a;
  }
}

// ---------------------------------------------------------------------------
// rowsum[b,i] = sum_j attn[b,i,j]
// ---------------------------------------------------------------------------
__global__ void rowsum_kernel(const float* __restrict__ attn, float* __restrict__ rowsum) {
  __shared__ float red[256];
  const int bi = blockIdx.x;
  const float* row = attn + (long)bi * N_;
  float s = 0.f;
  for (int j = threadIdx.x; j < N_; j += 256) s += row[j];
  red[threadIdx.x] = s;
  __syncthreads();
  for (int st = 128; st > 0; st >>= 1) {
    if (threadIdx.x < st) red[threadIdx.x] += red[threadIdx.x + st];
    __syncthreads();
  }
  if (!threadIdx.x) rowsum[bi] = red[0];
}

// ---------------------------------------------------------------------------
// updates[b,i,d] = sum_j (attn[b,i,j]/rowsum[b,i]) * v[b,j,d]   via WMMA.
// V is pre-transposed -> B-fragments are single contiguous 32B vector loads.
// Padded A rows (m>=8) zeroed through inv=0 -> no EXEC divergence at WMMA.
// ---------------------------------------------------------------------------
__global__ void updates_kernel(const float* __restrict__ attn, const float* __restrict__ rowsum,
                               const __bf16* __restrict__ v_t, float* __restrict__ updates) {
  const int b = blockIdx.x;
  const int tid = threadIdx.x, lane = tid & 31, wave = tid >> 5;
  const int m = lane & 15, kb = (lane >> 4) * 8, kb2 = (lane >> 4) * 16;
  const float inv = (m < NS_) ? 1.0f / rowsum[b * NS_ + m] : 0.0f;
  const float* arow = attn + ((long)b * NS_ + (m < NS_ ? m : 0)) * N_;
  for (int t = wave; t < 16; t += 8) {
    const int dcol = t * 16 + m;
    const __bf16* vcol = v_t + ((long)b * D_ + dcol) * N_;
    f32x8 acc = {};
    for (int kc = 0; kc < N_ / 32; ++kc) {
      const int k0 = kc * 32;
      bf16x16 a, bm;
#pragma unroll
      for (int p = 0; p < 8; ++p) {
        a[p]     = (__bf16)(arow[k0 + kb2 + p] * inv);
        a[8 + p] = (__bf16)(arow[k0 + kb2 + 8 + p] * inv);
      }
      bm = *(const bf16x16*)(vcol + k0 + kb2);
      acc = __builtin_amdgcn_wmma_f32_16x16x32_bf16(false, a, false, bm, (short)0, acc,
                                                    false, false);
    }
    if (lane < 16) {
#pragma unroll
      for (int r = 0; r < NS_; ++r)
        updates[((long)b * NS_ + r) * D_ + t * 16 + m] = acc[r];
    }
  }
}

// ---------------------------------------------------------------------------
// rel message MLP (tiny, VALU)
// ---------------------------------------------------------------------------
__global__ void rel_kernel(const float* __restrict__ slots_prev,
                           const float* __restrict__ rel_w1, const float* __restrict__ rel_b1,
                           const float* __restrict__ rel_w2, const float* __restrict__ rel_b2,
                           float* __restrict__ rel) {
  __shared__ float sp[NS_][D_];
  __shared__ float xa[NS_][HID_], xb[NS_][HID_];
  __shared__ float h[NS_][NS_][HID_];
  const int b = blockIdx.x, tid = threadIdx.x;
  for (int o = tid; o < NS_ * D_; o += 256)
    sp[o >> 8][o & 255] = slots_prev[(long)b * NS_ * D_ + o];
  __syncthreads();
  for (int o = tid; o < NS_ * HID_; o += 256) {
    const int i = o / HID_, hh = o % HID_;
    const float* w = rel_w1 + (long)hh * (2 * D_);
    float sa = 0.f, sb = 0.f;
    for (int d = 0; d < D_; ++d) { sa += sp[i][d] * w[d]; sb += sp[i][d] * w[D_ + d]; }
    xa[i][hh] = sa;
    xb[i][hh] = sb;
  }
  __syncthreads();
  for (int o = tid; o < NS_ * NS_ * HID_; o += 256) {
    const int i = o / (NS_ * HID_), j = (o / HID_) % NS_, hh = o % HID_;
    h[i][j][hh] = fmaxf(0.f, xa[i][hh] + xb[j][hh] + rel_b1[hh]);
  }
  __syncthreads();
  for (int o = tid; o < NS_ * D_; o += 256) {
    const int i = o >> 8, d = o & 255;
    const float* w2 = rel_w2 + (long)d * HID_;
    float s = 0.f;
    for (int j = 0; j < NS_; ++j) {
      if (j == i) continue;
      float ss = 0.f;
      for (int hh = 0; hh < HID_; ++hh) ss += h[i][j][hh] * w2[hh];
      s += ss;
    }
    rel[(long)b * NS_ * D_ + o] = s * (1.0f / (NS_ - 1)) + rel_b2[d];
  }
}

// ---------------------------------------------------------------------------
// Fused GRU cell + LN + FF residual per slot row (one block per (b, slot))
// ---------------------------------------------------------------------------
__global__ void gru_ff_kernel(const float* __restrict__ updates, const float* __restrict__ rel,
                              const float* __restrict__ slots_prev,
                              const float* __restrict__ w_ih, const float* __restrict__ w_hh,
                              const float* __restrict__ b_ih, const float* __restrict__ b_hh,
                              const float* __restrict__ ln_ff_g, const float* __restrict__ ln_ff_b,
                              const float* __restrict__ obj_w1, const float* __restrict__ obj_b1,
                              const float* __restrict__ obj_w2, const float* __restrict__ obj_b2,
                              float* __restrict__ slots) {
  __shared__ float gin[2 * D_], hprev[D_], gx[3 * D_], gh[3 * D_];
  __shared__ float snew[D_], red[256], ffx[D_], hid[HID_];
  const int ri = blockIdx.x, tid = threadIdx.x;
  for (int o = tid; o < D_; o += 256) {
    gin[o] = updates[(long)ri * D_ + o];
    gin[D_ + o] = rel[(long)ri * D_ + o];
    hprev[o] = slots_prev[(long)ri * D_ + o];
  }
  __syncthreads();
  for (int go = tid; go < 3 * D_; go += 256) {
    const float* wi = w_ih + (long)go * (2 * D_);
    const float* wh = w_hh + (long)go * D_;
    float sx = b_ih[go], sh = b_hh[go];
    for (int c = 0; c < 2 * D_; ++c) sx += gin[c] * wi[c];
    for (int c = 0; c < D_; ++c) sh += hprev[c] * wh[c];
    gx[go] = sx;
    gh[go] = sh;
  }
  __syncthreads();
  {
    const int d = tid;
    const float r = 1.0f / (1.0f + __expf(-(gx[d] + gh[d])));
    const float z = 1.0f / (1.0f + __expf(-(gx[D_ + d] + gh[D_ + d])));
    const float n = tanhf(gx[2 * D_ + d] + r * gh[2 * D_ + d]);
    snew[d] = (1.0f - z) * n + z * hprev[d];
  }
  __syncthreads();
  red[tid] = snew[tid];
  __syncthreads();
  for (int st = 128; st > 0; st >>= 1) {
    if (tid < st) red[tid] += red[tid + st];
    __syncthreads();
  }
  const float mu = red[0] * (1.0f / D_);
  __syncthreads();
  {
    const float d0 = snew[tid] - mu;
    red[tid] = d0 * d0;
  }
  __syncthreads();
  for (int st = 128; st > 0; st >>= 1) {
    if (tid < st) red[tid] += red[tid + st];
    __syncthreads();
  }
  const float inv = rsqrtf(red[0] * (1.0f / D_) + 1e-5f);
  ffx[tid] = (snew[tid] - mu) * inv * ln_ff_g[tid] + ln_ff_b[tid];
  __syncthreads();
  if (tid < HID_) {
    const float* w1 = obj_w1 + (long)tid * D_;
    float s = obj_b1[tid];
    for (int d = 0; d < D_; ++d) s += ffx[d] * w1[d];
    hid[tid] = fmaxf(0.f, s);
  }
  __syncthreads();
  {
    const float* w2 = obj_w2 + (long)tid * HID_;
    float s = obj_b2[tid];
    for (int hh = 0; hh < HID_; ++hh) s += hid[hh] * w2[hh];
    slots[(long)ri * D_ + tid] = snew[tid] + s;
  }
}

__global__ void copy_kernel(const float* __restrict__ src, float* __restrict__ dst, int n) {
  const int i = blockIdx.x * 256 + threadIdx.x;
  if (i < n) dst[i] = src[i];
}

// ---------------------------------------------------------------------------
extern "C" void kernel_launch(void* const* d_in, const int* in_sizes, int n_in,
                              void* d_out, int out_size, void* d_ws, size_t ws_size,
                              hipStream_t stream) {
  const float* inputs  = (const float*)d_in[0];
  const float* noise   = (const float*)d_in[1];
  const float* smu     = (const float*)d_in[2];
  const float* ssig    = (const float*)d_in[3];
  const float* Wq      = (const float*)d_in[4];
  const float* Wk      = (const float*)d_in[5];
  const float* Wv      = (const float*)d_in[6];
  const float* w_ih    = (const float*)d_in[7];
  const float* w_hh    = (const float*)d_in[8];
  const float* b_ih    = (const float*)d_in[9];
  const float* b_hh    = (const float*)d_in[10];
  const float* obj_w1  = (const float*)d_in[11];
  const float* obj_b1  = (const float*)d_in[12];
  const float* obj_w2  = (const float*)d_in[13];
  const float* obj_b2  = (const float*)d_in[14];
  const float* rel_w1  = (const float*)d_in[15];
  const float* rel_b1  = (const float*)d_in[16];
  const float* rel_w2  = (const float*)d_in[17];
  const float* rel_b2  = (const float*)d_in[18];
  const float* ln_in_g = (const float*)d_in[19];
  const float* ln_in_b = (const float*)d_in[20];
  const float* ln_s_g  = (const float*)d_in[21];
  const float* ln_s_b  = (const float*)d_in[22];
  const float* ln_ff_g = (const float*)d_in[23];
  const float* ln_ff_b = (const float*)d_in[24];

  char* ws = (char*)d_ws;
  size_t off = 0;
  auto alloc = [&](size_t bytes) -> char* {
    char* p = ws + off;
    off += (bytes + 255) & ~(size_t)255;
    return p;
  };
  __bf16* wk_bf    = (__bf16*)alloc((size_t)D_ * D_ * 2);
  __bf16* wv_bf    = (__bf16*)alloc((size_t)D_ * D_ * 2);
  float* slots     = (float*)alloc((size_t)B_ * NS_ * D_ * 4);
  float* slots_prev= (float*)alloc((size_t)B_ * NS_ * D_ * 4);
  __bf16* q_bf     = (__bf16*)alloc((size_t)B_ * 16 * D_ * 2);
  float* dots      = (float*)alloc((size_t)B_ * NS_ * N_ * 4);
  float* rowsum    = (float*)alloc((size_t)B_ * NS_ * 4);
  float* updates   = (float*)alloc((size_t)B_ * NS_ * D_ * 4);
  float* rel       = (float*)alloc((size_t)B_ * NS_ * D_ * 4);
  __bf16* k_bf     = (__bf16*)alloc((size_t)B_ * N_ * D_ * 2);
  __bf16* v_t      = (__bf16*)alloc((size_t)B_ * N_ * D_ * 2);

  float* out_slots = (float*)d_out;
  float* out_attn  = out_slots + (size_t)B_ * NS_ * D_;

  const int prep_n = 2 * D_ * D_ + B_ * NS_ * D_;
  prep_kernel<<<(prep_n + 255) / 256, 256, 0, stream>>>(Wk, Wv, smu, ssig, noise, wk_bf,
                                                        wv_bf, slots);

  ln_kv_kernel<<<(B_ * N_) / 16, 256, 0, stream>>>(inputs, ln_in_g, ln_in_b, wk_bf, wv_bf,
                                                   k_bf, v_t);

  for (int it = 0; it < ITERS_; ++it) {
    copy_kernel<<<(B_ * NS_ * D_) / 256, 256, 0, stream>>>(slots, slots_prev, B_ * NS_ * D_);
    qproj_kernel<<<B_, 256, 0, stream>>>(slots, ln_s_g, ln_s_b, Wq, q_bf);
    dots_kernel<<<B_ * 32, 256, 0, stream>>>(q_bf, k_bf, dots);
    softmax_kernel<<<(B_ * N_) / 256, 256, 0, stream>>>(
        dots, out_attn + (size_t)it * B_ * NS_ * N_);
    rowsum_kernel<<<B_ * NS_, 256, 0, stream>>>(dots, rowsum);
    updates_kernel<<<B_, 256, 0, stream>>>(dots, rowsum, v_t, updates);
    rel_kernel<<<B_, 256, 0, stream>>>(slots_prev, rel_w1, rel_b1, rel_w2, rel_b2, rel);
    gru_ff_kernel<<<B_ * NS_, 256, 0, stream>>>(updates, rel, slots_prev, w_ih, w_hh, b_ih,
                                                b_hh, ln_ff_g, ln_ff_b, obj_w1, obj_b1,
                                                obj_w2, obj_b2, slots);
  }
  copy_kernel<<<(B_ * NS_ * D_) / 256, 256, 0, stream>>>(slots, out_slots, B_ * NS_ * D_);
}
